// Attention_29953101922931
// MI455X (gfx1250) — compile-verified, matrix operand order checked by reading, not run on hardware
//
#include <hip/hip_runtime.h>
#include <hip/hip_bf16.h>

#define BB 8
#define SS 1024
#define HH 16
#define DD 64
#define WW 1024   // HH*DD

typedef __attribute__((ext_vector_type(16))) __bf16 v16bf;
typedef __attribute__((ext_vector_type(4)))  __bf16 v4bf;
typedef __attribute__((ext_vector_type(8)))  float  v8f;

union FragBF {
  v16bf v;
  uint4 q[2];
};

// Native fptrunc -> v_cvt_pk_bf16_f32 on gfx1250.
__device__ inline __bf16 f2bf(float f) { return (__bf16)f; }

__device__ inline v8f wmma_bf16(const FragBF& a, const FragBF& b, v8f c) {
  return __builtin_amdgcn_wmma_f32_16x16x32_bf16(
      /*neg_a=*/false, a.v, /*neg_b=*/false, b.v,
      /*c_mod=*/(short)0, c, /*reuse_a=*/false, /*reuse_b=*/false);
}

// ---------------------------------------------------------------------------
// Kernel 1: fused QKV projection.  C = A @ W + bias, fp32 in, bf16 out.
// blockIdx.z: 0 -> Q (A=from), 1 -> K (A=to), 2 -> V (A=to, stored transposed)
// Block: 256 threads (8 waves). Output tile 128x64: wave w owns row tile w
// (16 rows) x all 4 col tiles -> 4 WMMA per wave per K-step, A-frag reused.
// Global loads for tile k+1 are software-pipelined in registers so HBM
// latency overlaps the WMMA phase.
// ---------------------------------------------------------------------------
__global__ __launch_bounds__(256) void qkv_proj_kernel(
    const float* __restrict__ from_t, const float* __restrict__ to_t,
    const float* __restrict__ Wq, const float* __restrict__ bq,
    const float* __restrict__ Wk, const float* __restrict__ bk,
    const float* __restrict__ Wv, const float* __restrict__ bv,
    __bf16* __restrict__ q_ws,
    __bf16* __restrict__ k_ws,
    __bf16* __restrict__ vt_ws)
{
  const int which = blockIdx.z;
  const float* Amat = (which == 0) ? from_t : to_t;
  const float* Wmat = (which == 0) ? Wq : (which == 1) ? Wk : Wv;
  const float* bias = (which == 0) ? bq : (which == 1) ? bk : bv;

  const int m0 = blockIdx.x * 128;  // row block in [0, 8192)
  const int n0 = blockIdx.y * 64;   // col block in [0, 1024)

  __shared__ __bf16 lds_a[128][32];  // [m][k]        8 KB
  __shared__ __bf16 lds_b[64][32];   // [n][k] (W^T)  4 KB

  const int tid  = threadIdx.x;
  const int lane = tid & 31;
  const int wave = tid >> 5;          // 0..7 = row tile
  const int half = lane >> 4;         // 0/1
  const int l15  = lane & 15;

  // Per-thread staging coordinates (A: 4 float4, W: 2 float4)
  int arow[4], acol[4];
  #pragma unroll
  for (int i = 0; i < 4; ++i) {
    int c = tid + i * 256;            // 0..1023
    arow[i] = c >> 3;                 // 8 float4 per 32-wide row
    acol[i] = (c & 7) << 2;
  }
  int wk[2], wn[2];
  #pragma unroll
  for (int i = 0; i < 2; ++i) {
    int c = tid + i * 256;            // 0..511
    wk[i] = c >> 4;                   // 16 float4 per k-row
    wn[i] = (c & 15) << 2;
  }

  v8f acc[4];
  const v8f zero8 = {0.f,0.f,0.f,0.f,0.f,0.f,0.f,0.f};
  #pragma unroll
  for (int nt = 0; nt < 4; ++nt) acc[nt] = zero8;

  // ---- prologue: load k0 = 0 tile into registers
  float4 fa[4], fw[2];
  #pragma unroll
  for (int i = 0; i < 4; ++i)
    fa[i] = *(const float4*)(Amat + (size_t)(m0 + arow[i]) * WW + acol[i]);
  #pragma unroll
  for (int i = 0; i < 2; ++i)
    fw[i] = *(const float4*)(Wmat + (size_t)wk[i] * WW + n0 + wn[i]);

  for (int k0 = 0; k0 < WW; k0 += 32) {
    // ---- convert + store the in-flight tile to LDS
    #pragma unroll
    for (int i = 0; i < 4; ++i) {
      v4bf p = {f2bf(fa[i].x), f2bf(fa[i].y), f2bf(fa[i].z), f2bf(fa[i].w)};
      *(v4bf*)&lds_a[arow[i]][acol[i]] = p;
    }
    #pragma unroll
    for (int i = 0; i < 2; ++i) {
      lds_b[wn[i] + 0][wk[i]] = f2bf(fw[i].x);
      lds_b[wn[i] + 1][wk[i]] = f2bf(fw[i].y);
      lds_b[wn[i] + 2][wk[i]] = f2bf(fw[i].z);
      lds_b[wn[i] + 3][wk[i]] = f2bf(fw[i].w);
    }
    __syncthreads();

    // ---- issue next tile's global loads (overlap with WMMA phase)
    const int k1 = k0 + 32;
    if (k1 < WW) {
      #pragma unroll
      for (int i = 0; i < 4; ++i)
        fa[i] = *(const float4*)(Amat + (size_t)(m0 + arow[i]) * WW + k1 + acol[i]);
      #pragma unroll
      for (int i = 0; i < 2; ++i)
        fw[i] = *(const float4*)(Wmat + (size_t)(k1 + wk[i]) * WW + n0 + wn[i]);
    }

    // ---- fragments + 4 WMMA (A fragment reused across 4 B tiles)
    FragBF a;
    const __bf16* ap = &lds_a[wave * 16 + l15][half * 8];
    a.q[0] = *(const uint4*)(ap);
    a.q[1] = *(const uint4*)(ap + 16);
    #pragma unroll
    for (int nt = 0; nt < 4; ++nt) {
      FragBF b;
      const __bf16* bp = &lds_b[nt * 16 + l15][half * 8];
      b.q[0] = *(const uint4*)(bp);
      b.q[1] = *(const uint4*)(bp + 16);
      acc[nt] = wmma_bf16(a, b, acc[nt]);
    }
    __syncthreads();
  }

  // ---- epilogue: bias, bf16 convert, scatter into attention layouts
  #pragma unroll
  for (int nt = 0; nt < 4; ++nt) {
    const int n = n0 + nt * 16 + l15;
    const float bs = bias[n];
    const int h = n >> 6;
    const int d = n & 63;
    #pragma unroll
    for (int r = 0; r < 8; ++r) {
      int m = m0 + wave * 16 + half * 8 + r;   // C layout: row = r + 8*half
      int bidx = m >> 10;
      int s    = m & 1023;
      float v = acc[nt][r] + bs;
      if (which == 2) {
        // V stored transposed: [B,H,D,S]
        vt_ws[(((size_t)bidx * HH + h) * DD + d) * SS + s] = f2bf(v);
      } else {
        __bf16* dst = (which == 0) ? q_ws : k_ws;   // [B,H,S,D]
        dst[(((size_t)bidx * HH + h) * SS + s) * DD + d] = f2bf(v);
      }
    }
  }
}

// ---------------------------------------------------------------------------
// Kernel 2: flash attention. One wave per 16-query tile; online softmax over
// 64 key tiles processed in pairs (K=32 for the P@V WMMA). 8 waves per block.
// ---------------------------------------------------------------------------
__global__ __launch_bounds__(256) void attn_kernel(
    const __bf16* __restrict__ q_ws,
    const __bf16* __restrict__ k_ws,
    const __bf16* __restrict__ vt_ws,
    float* __restrict__ out)
{
  const int bh   = blockIdx.x;            // b*H + h
  const int wave = threadIdx.x >> 5;
  const int lane = threadIdx.x & 31;
  const int qt   = blockIdx.y * 8 + wave; // query tile 0..63
  const int half = lane >> 4;
  const int l15  = lane & 15;

  const __bf16* qb = q_ws  + (size_t)bh * SS * DD;
  const __bf16* kb = k_ws  + (size_t)bh * SS * DD;
  const __bf16* vb = vt_ws + (size_t)bh * DD * SS;

  __shared__ __bf16 lds_p[8][16][32];   // per-wave P staging (C->A relayout)

  // Q fragments for d=[0,32) and [32,64); resident for the whole kernel.
  FragBF aq0, aq1;
  {
    const __bf16* p = qb + (size_t)(qt * 16 + l15) * DD + half * 8;
    aq0.q[0] = *(const uint4*)(p);
    aq0.q[1] = *(const uint4*)(p + 16);
    aq1.q[0] = *(const uint4*)(p + 32);
    aq1.q[1] = *(const uint4*)(p + 48);
  }

  const float sl2e = 0.125f * 1.44269504088896341f;  // 1/sqrt(64) * log2(e)

  float m_i[8], l_i[8];
  #pragma unroll
  for (int r = 0; r < 8; ++r) { m_i[r] = -1e30f; l_i[r] = 0.f; }
  const v8f zero8 = {0.f,0.f,0.f,0.f,0.f,0.f,0.f,0.f};
  v8f ctx[4] = {zero8, zero8, zero8, zero8};

  for (int j = 0; j < SS / 32; ++j) {
    // prefetch next iteration's K rows and V rows (global_prefetch_b8)
    if (j + 1 < SS / 32) {
      __builtin_prefetch(kb + (size_t)((2 * j + 2) * 16 + l15) * DD, 0, 3);
      __builtin_prefetch(vb + (size_t)l15 * SS + 32 * (j + 1), 0, 3);
    }

    // ---- scores for key tiles 2j and 2j+1:  S = Q @ K^T (K rows are B columns)
    v8f s0 = zero8, s1 = zero8;
    {
      FragBF bk0, bk1;
      const __bf16* p = kb + (size_t)((2*j + 0) * 16 + l15) * DD + half * 8;
      bk0.q[0] = *(const uint4*)(p);
      bk0.q[1] = *(const uint4*)(p + 16);
      bk1.q[0] = *(const uint4*)(p + 32);
      bk1.q[1] = *(const uint4*)(p + 48);
      s0 = wmma_bf16(aq0, bk0, s0);
      s0 = wmma_bf16(aq1, bk1, s0);
    }
    {
      FragBF bk0, bk1;
      const __bf16* p = kb + (size_t)((2*j + 1) * 16 + l15) * DD + half * 8;
      bk0.q[0] = *(const uint4*)(p);
      bk0.q[1] = *(const uint4*)(p + 16);
      bk1.q[0] = *(const uint4*)(p + 32);
      bk1.q[1] = *(const uint4*)(p + 48);
      s1 = wmma_bf16(aq0, bk0, s1);
      s1 = wmma_bf16(aq1, bk1, s1);
    }

    float t0[8], t1[8];
    #pragma unroll
    for (int r = 0; r < 8; ++r) { t0[r] = s0[r] * sl2e; t1[r] = s1[r] * sl2e; }

    // ---- online softmax per row (rows split at lane 16; xor masks 1..8 stay in-half)
    float alpha[8];
    #pragma unroll
    for (int r = 0; r < 8; ++r) {
      float x = fmaxf(t0[r], t1[r]);
      #pragma unroll
      for (int off = 1; off < 16; off <<= 1)
        x = fmaxf(x, __shfl_xor(x, off, 32));
      float mnew = fmaxf(m_i[r], x);
      alpha[r]  = __builtin_exp2f(m_i[r] - mnew);
      float p0  = __builtin_exp2f(t0[r] - mnew);
      float p1  = __builtin_exp2f(t1[r] - mnew);
      float psum = p0 + p1;
      #pragma unroll
      for (int off = 1; off < 16; off <<= 1)
        psum += __shfl_xor(psum, off, 32);
      l_i[r] = l_i[r] * alpha[r] + psum;
      m_i[r] = mnew;
      // stage P (C layout row = r + 8*half, col = l15) into [16][32] row-major
      lds_p[wave][half * 8 + r][l15]      = f2bf(p0);
      lds_p[wave][half * 8 + r][16 + l15] = f2bf(p1);
      #pragma unroll
      for (int nt = 0; nt < 4; ++nt) ctx[nt][r] *= alpha[r];
    }

    // wave-local LDS RAW fence: drain DS counter, block compiler reordering
    asm volatile("s_wait_dscnt 0" ::: "memory");
    __builtin_amdgcn_wave_barrier();

    // ---- P as A-fragment (16x32 bf16)
    FragBF pf;
    const __bf16* pp = &lds_p[wave][l15][half * 8];
    pf.q[0] = *(const uint4*)(pp);
    pf.q[1] = *(const uint4*)(pp + 16);

    // ---- ctx += P @ V ; V^T layout makes B columns (d) row-contiguous
    #pragma unroll
    for (int nt = 0; nt < 4; ++nt) {
      FragBF bv;
      const __bf16* vp = vb + (size_t)(nt * 16 + l15) * SS + 32 * j + half * 8;
      bv.q[0] = *(const uint4*)(vp);
      bv.q[1] = *(const uint4*)(vp + 16);
      ctx[nt] = wmma_bf16(pf, bv, ctx[nt]);
    }
    __builtin_amdgcn_wave_barrier();
  }

  // ---- epilogue: 1/l scaling, write fp32 [B,S,H*D]
  const int b = bh >> 4;
  const int h = bh & 15;
  #pragma unroll
  for (int r = 0; r < 8; ++r) {
    float inv = 1.0f / l_i[r];
    int s = qt * 16 + half * 8 + r;
    float* op = out + ((size_t)b * SS + s) * WW + h * DD;
    #pragma unroll
    for (int nt = 0; nt < 4; ++nt)
      op[nt * 16 + l15] = ctx[nt][r] * inv;
  }
}

extern "C" void kernel_launch(void* const* d_in, const int* in_sizes, int n_in,
                              void* d_out, int out_size, void* d_ws, size_t ws_size,
                              hipStream_t stream) {
  const float* from_t = (const float*)d_in[0];
  const float* to_t   = (const float*)d_in[1];
  const float* Wq     = (const float*)d_in[2];
  const float* bq     = (const float*)d_in[3];
  const float* Wk     = (const float*)d_in[4];
  const float* bk     = (const float*)d_in[5];
  const float* Wv     = (const float*)d_in[6];
  const float* bv     = (const float*)d_in[7];

  const size_t elems = (size_t)BB * SS * WW;       // 8M bf16 per tensor
  __bf16* q_ws  = (__bf16*)d_ws;
  __bf16* k_ws  = q_ws + elems;
  __bf16* vt_ws = k_ws + elems;                    // V stored [B,H,D,S]

  qkv_proj_kernel<<<dim3(BB * SS / 128, WW / 64, 3), 256, 0, stream>>>(
      from_t, to_t, Wq, bq, Wk, bk, Wv, bv, q_ws, k_ws, vt_ws);

  attn_kernel<<<dim3(BB * HH, SS / (16 * 8), 1), 256, 0, stream>>>(
      q_ws, k_ws, vt_ws, (float*)d_out);
}
